// RnnModel_23012434772371
// MI455X (gfx1250) — matrix operand order repaired and checked by reading, not compiled, not run on hardware
//
#include <hip/hip_runtime.h>
#include <math.h>
#include <stdint.h>

typedef __attribute__((ext_vector_type(16))) __bf16 v16bf;
typedef __attribute__((ext_vector_type(8)))  float  v8f;
typedef int v4i __attribute__((vector_size(16)));   // matches builtin param type

#define TSTEPS 28
#define IN_DIM 256
#define HID    2048
#define BATCH  2048
#define KTOT   (IN_DIM + HID)   // 2304

#define BM 128
#define BN 128
#define KC 64                   // two WMMA K-steps per LDS stage
#define WM 32                   // per-wave M tile
#define WN 64                   // per-wave N tile
#define FRAG_M (WM / 16)        // 2
#define FRAG_N (WN / 16)        // 4
#define KITER  (KTOT / KC)      // 36
#define CH_PER_TILE (BM * KC / 8)   // 1024 x 16B chunks per tile

#define AS1 __attribute__((address_space(1)))
#define AS3 __attribute__((address_space(3)))

#if defined(__AMDGCN__) && __has_builtin(__builtin_amdgcn_global_load_async_to_lds_b128) && __has_builtin(__builtin_amdgcn_s_wait_asynccnt)
#define USE_ASYNC 1
#else
#define USE_ASYNC 0
#endif

// ---------- helpers ----------
__device__ __forceinline__ unsigned short f2bf(float f) {
  union { float f; unsigned u; } v; v.f = f;
  unsigned r = v.u + 0x7FFFu + ((v.u >> 16) & 1u);   // round-to-nearest-even
  return (unsigned short)(r >> 16);
}
__device__ __forceinline__ float bf2f(unsigned short b) {
  union { unsigned u; float f; } v; v.u = ((unsigned)b) << 16; return v.f;
}
__device__ __forceinline__ v16bf mkfrag(uint4 lo, uint4 hi) {
  union { uint4 u[2]; v16bf v; } x; x.u[0] = lo; x.u[1] = hi; return x.v;
}
__device__ __forceinline__ void copy16_async(const uint4* g, uint4* l) {
#if USE_ASYNC
  __builtin_amdgcn_global_load_async_to_lds_b128(
      (AS1 v4i*)(uintptr_t)g, (AS3 v4i*)(uintptr_t)l, 0, 0);
#else
  *l = *g;
#endif
}

// ---------- setup kernels ----------
__global__ void cvt_f32_bf16(const float* __restrict__ in,
                             unsigned short* __restrict__ out, int n) {
  int i = blockIdx.x * blockDim.x + threadIdx.x;
  if (i < n) out[i] = f2bf(in[i]);
}

// W_cell: [KTOT][HID] row-major fp32  ->  Wt: [HID][KTOT] bf16 (N-major)
__global__ void transpose_w(const float* __restrict__ W,
                            unsigned short* __restrict__ Wt) {
  int n = blockIdx.x * blockDim.x + threadIdx.x;  // 0..HID-1
  int k = blockIdx.y;                             // 0..KTOT-1
  Wt[(size_t)n * KTOT + k] = f2bf(W[(size_t)k * HID + n]);
}

__global__ void zero_u32(unsigned int* __restrict__ p, int n) {
  int i = blockIdx.x * blockDim.x + threadIdx.x;
  if (i < n) p[i] = 0u;
}

// ---------- one recurrence step: hout = tanh([x_t | hin] @ W + b) ----------
__global__ __launch_bounds__(256) void rnn_step(
    const unsigned short* __restrict__ xbf,   // [BATCH][TSTEPS][IN_DIM] bf16
    const unsigned short* __restrict__ Wt,    // [HID][KTOT] bf16 (B^T)
    const float* __restrict__ bcell,          // [HID]
    const unsigned short* __restrict__ hin,   // [BATCH][HID] bf16
    unsigned short* __restrict__ hout,        // [BATCH][HID] bf16
    int t)
{
  // double-buffered tiles: A [BM][KC] bf16, B(^T) [BN][KC] bf16 ; 64 KB total
  __shared__ uint4 sA[2][CH_PER_TILE];
  __shared__ uint4 sB[2][CH_PER_TILE];

  const int tid  = threadIdx.x;
  const int lane = tid & 31;
  const int wave = tid >> 5;
  const int r    = lane & 15;         // row/col within fragment
  const int hh   = lane >> 4;         // half-wave select
  const int bm   = blockIdx.x * BM;   // batch offset
  const int bn   = blockIdx.y * BN;   // hidden (output) offset
  const int wm   = (wave & 3) * WM;   // 4 waves along M
  const int wn   = (wave >> 2) * WN;  // 2 waves along N

  v8f acc[FRAG_M][FRAG_N] = {};

  // Issue one K-stage of async fills (8 x b128 per thread: 4 for A, 4 for B).
  // A source is wave-uniform per stage (k0 < 256 -> x_t, else -> h); 256 % KC == 0.
  auto issueStage = [&](int buf, int kc) {
    const int k0 = kc * KC;
    const unsigned short* abase;
    size_t astride;
    if (k0 < IN_DIM) {
      abase   = xbf + ((size_t)bm * TSTEPS + t) * IN_DIM + k0;
      astride = (size_t)TSTEPS * IN_DIM;
    } else {
      abase   = hin + (size_t)bm * HID + (k0 - IN_DIM);
      astride = HID;
    }
    const unsigned short* bbase = Wt + (size_t)bn * KTOT + k0;
    #pragma unroll
    for (int q = 0; q < 4; ++q) {
      int c   = tid + q * 256;        // chunk id 0..1023
      int row = c >> 3;               // 0..127 (row of A / col of W^T)
      int seg = c & 7;                // which 8-bf16 group along K
      copy16_async(
          reinterpret_cast<const uint4*>(abase + (size_t)row * astride + seg * 8),
          &sA[buf][c]);
      copy16_async(
          reinterpret_cast<const uint4*>(bbase + (size_t)row * KTOT + seg * 8),
          &sB[buf][c]);
    }
  };

  issueStage(0, 0);
  issueStage(1, 1);

  int buf = 0;
  for (int kc = 0; kc < KITER; ++kc) {
#if USE_ASYNC
    if (kc + 1 < KITER) __builtin_amdgcn_s_wait_asynccnt(8);  // oldest 8 (= this stage) done
    else                __builtin_amdgcn_s_wait_asynccnt(0);
#endif
    __syncthreads();   // stage `kc` visible to all waves

    #pragma unroll
    for (int s = 0; s < 2; ++s) {     // two WMMA K-steps per stage
      v16bf af[FRAG_M], bfv[FRAG_N];
      // A 16x32 bf16 frag: lane holds M=r, K=[hh*16..+15]
      #pragma unroll
      for (int i = 0; i < FRAG_M; ++i) {
        int row = wm + i * 16 + r;
        af[i] = mkfrag(sA[buf][row * 8 + s * 4 + hh * 2],
                       sA[buf][row * 8 + s * 4 + hh * 2 + 1]);
      }
      // B 32x16 bf16 frag from [N][K] tile: lane holds col N=r, K=[hh*16..+15]
      #pragma unroll
      for (int j = 0; j < FRAG_N; ++j) {
        int col = wn + j * 16 + r;
        bfv[j] = mkfrag(sB[buf][col * 8 + s * 4 + hh * 2],
                        sB[buf][col * 8 + s * 4 + hh * 2 + 1]);
      }
      #pragma unroll
      for (int i = 0; i < FRAG_M; ++i)
        #pragma unroll
        for (int j = 0; j < FRAG_N; ++j)
          acc[i][j] = __builtin_amdgcn_wmma_f32_16x16x32_bf16(
              false, af[i], false, bfv[j], (short)0, acc[i][j], false, false);
    }

    __syncthreads();   // all waves done reading `buf` before it is refilled
    if (kc + 2 < KITER) issueStage(buf, kc + 2);
    buf ^= 1;
  }

  // epilogue: bias + tanh + bf16 store. C layout: M = vgpr + 8*hh, N = lane&15
  #pragma unroll
  for (int j = 0; j < FRAG_N; ++j) {
    int col = bn + wn + j * 16 + r;
    float bc = bcell[col];
    #pragma unroll
    for (int i = 0; i < FRAG_M; ++i) {
      int row0 = bm + wm + i * 16 + hh * 8;
      #pragma unroll
      for (int v = 0; v < 8; ++v) {
        float val = acc[i][j][v] + bc;
        hout[(size_t)(row0 + v) * HID + col] = f2bf(tanhf(val));
      }
    }
  }
}

// ---------- classifier head: out = h_last @ W_out + b_out (N=10, plain VALU) ----------
__global__ __launch_bounds__(256) void head_kernel(
    const unsigned short* __restrict__ h,     // [BATCH][HID] bf16
    const float* __restrict__ Wout,           // [HID][10]
    const float* __restrict__ bout,           // [10]
    float* __restrict__ out)                  // [BATCH][10]
{
  __shared__ float red[256];
  const int m   = blockIdx.x;
  const int tid = threadIdx.x;

  float acc[10] = {0.f};
  uint4 hv4 = *reinterpret_cast<const uint4*>(h + (size_t)m * HID + tid * 8);
  const unsigned short* hs = reinterpret_cast<const unsigned short*>(&hv4);
  #pragma unroll
  for (int v = 0; v < 8; ++v) {
    float hv = bf2f(hs[v]);
    const float* w = Wout + (size_t)(tid * 8 + v) * 10;
    #pragma unroll
    for (int c = 0; c < 10; ++c) acc[c] += hv * w[c];
  }
  for (int c = 0; c < 10; ++c) {
    red[tid] = acc[c];
    __syncthreads();
    for (int s = 128; s > 0; s >>= 1) {
      if (tid < s) red[tid] += red[tid + s];
      __syncthreads();
    }
    if (tid == 0) out[m * 10 + c] = red[0] + bout[c];
    __syncthreads();
  }
}

// ---------- launcher ----------
extern "C" void kernel_launch(void* const* d_in, const int* in_sizes, int n_in,
                              void* d_out, int out_size, void* d_ws, size_t ws_size,
                              hipStream_t stream) {
  (void)in_sizes; (void)n_in; (void)out_size; (void)ws_size;
  const float* x     = (const float*)d_in[0];   // [2048][28][256]
  const float* Wcell = (const float*)d_in[1];   // [2304][2048]
  const float* bcell = (const float*)d_in[2];   // [2048]
  const float* Wout  = (const float*)d_in[3];   // [2048][10]
  const float* bout  = (const float*)d_in[4];   // [10]
  float* out = (float*)d_out;

  char* ws = (char*)d_ws;
  const size_t WT_OFF  = 0;                              // 2048*2304*2 = 9,437,184
  const size_t XBF_OFF = 9437184;                        // 2048*28*256*2 = 29,360,128
  const size_t HA_OFF  = 38797312;                       // 2048*2048*2 = 8,388,608
  const size_t HB_OFF  = 47185920;                       // 8,388,608  (total ~55.6 MB)
  unsigned short* Wt  = (unsigned short*)(ws + WT_OFF);
  unsigned short* xbf = (unsigned short*)(ws + XBF_OFF);
  unsigned short* hA  = (unsigned short*)(ws + HA_OFF);
  unsigned short* hB  = (unsigned short*)(ws + HB_OFF);

  // setup: convert x, transpose+convert W, zero h0
  const int NX = BATCH * TSTEPS * IN_DIM;                // 14,680,064
  cvt_f32_bf16<<<(NX + 255) / 256, 256, 0, stream>>>(x, xbf, NX);
  transpose_w<<<dim3(HID / 256, KTOT), 256, 0, stream>>>(Wcell, Wt);
  const int NH32 = BATCH * HID / 2;                      // bf16 pairs as u32
  zero_u32<<<(NH32 + 255) / 256, 256, 0, stream>>>((unsigned int*)hA, NH32);

  // 28 dependent GEMM+tanh steps, double-buffered h
  unsigned short* cur = hA;
  unsigned short* nxt = hB;
  for (int t = 0; t < TSTEPS; ++t) {
    rnn_step<<<dim3(BATCH / BM, HID / BN), 256, 0, stream>>>(
        xbf, Wt, bcell, cur, nxt, t);
    unsigned short* tmp = cur; cur = nxt; nxt = tmp;
  }

  // classifier head on h_last (== hA after 28 swaps)
  head_kernel<<<BATCH, 256, 0, stream>>>(cur, Wout, bout, out);
}
